// PredatorCritic_81338090652058
// MI455X (gfx1250) — matrix operand order repaired and checked by reading, not compiled, not run on hardware
//
#include <hip/hip_runtime.h>
#include <hip/hip_bf16.h>

#define N_NODES 20000
#define N_EDGES 50000
#define P_PRED  5000
#define H       64
#define HH      4096           // H*H
#define NEGS    0.1f

typedef __attribute__((ext_vector_type(16))) _Float16 v16h;
typedef __attribute__((ext_vector_type(8)))  float    v8f;

__device__ __forceinline__ float lrelu(float x) { return x >= 0.f ? x : NEGS * x; }

// ---------------------------------------------------------------------------
// Pack W2 (64 x 4096, f32 row-major) into f16 B-fragments: each lane's v16h
// WMMA B operand is one contiguous 32-byte chunk.  B layout used here
// (16-bit B, 32x16): N = lane%16; lanes 0-15 hold K=0..15 (2/VGPR),
// lanes 16-31 hold K=16..31.   Index: W2h[((tile*2+kb)*32 + lane)*16 + half]
// ---------------------------------------------------------------------------
__global__ void pack_w2_kernel(const float* __restrict__ W2, _Float16* __restrict__ W2h) {
    int id   = blockIdx.x * 256 + threadIdx.x;      // 256 tiles * 2 kb * 32 lanes * 16
    int hh   = id & 15;
    int lane = (id >> 4) & 31;
    int kb   = (id >> 9) & 1;
    int tile = id >> 10;
    int v    = hh >> 1, p = hh & 1;
    int K    = 2 * v + p + ((lane < 16) ? 0 : 16);
    int k    = kb * 32 + K;
    int n    = tile * 16 + (lane & 15);
    W2h[id]  = (_Float16)W2[k * HH + n];
}

// ---------------------------------------------------------------------------
// Node embedding: h = lrelu(lrelu(x@W0+b0)@W1+b1) + ent_emb[mask] (+ a, n<P)
// ---------------------------------------------------------------------------
__global__ void node_kernel(const float* __restrict__ x, const float* __restrict__ action,
                            const int* __restrict__ mask,
                            const float* __restrict__ W0, const float* __restrict__ B0,
                            const float* __restrict__ W1, const float* __restrict__ B1,
                            const float* __restrict__ entEmb,
                            const float* __restrict__ Wa, const float* __restrict__ Ba,
                            float* __restrict__ hbuf, float* __restrict__ abuf) {
    int n = blockIdx.x * blockDim.x + threadIdx.x;
    if (n >= N_NODES) return;
    float x0 = x[n * 3 + 0], x1 = x[n * 3 + 1], x2 = x[n * 3 + 2];
    float h1[H];
    for (int i = 0; i < H; ++i)
        h1[i] = lrelu(x0 * W0[i] + x1 * W0[H + i] + x2 * W0[2 * H + i] + B0[i]);
    int   m  = mask[n];
    float av = (n < P_PRED) ? action[n] : 0.f;
    for (int i = 0; i < H; ++i) {
        float acc = B1[i];
        for (int k = 0; k < H; ++k) acc += h1[k] * W1[k * H + i];
        float h2 = lrelu(acc) + entEmb[m * H + i];
        if (n < P_PRED) {
            float a = lrelu(av * Wa[i] + Ba[i]);
            abuf[n * H + i] = a;
            h2 += a;
        }
        hbuf[n * H + i] = h2;
    }
}

// ---------------------------------------------------------------------------
// Fused edge kernel: 16 edges / block, 128 threads (4 waves).
// ---------------------------------------------------------------------------
#define PITCH   4104               // 4096 + 8 floats: conflict-free strided reads
#define GBPITCH 66                 // float2 pitch for transposed (g2,be2) table
// dynamic LDS layout (bytes)
#define OFF_SC   0                 // 16*4104*4            = 262656
#define OFF_SGB  262656            // 64*66*8 (float2)     = 33792
#define OFF_SH   296448            // 16*64*4
#define OFF_PS1  300544            // 4*32*8*4 partial S1  = 4096
#define OFF_PS2  304640            // 4*32*8*4 partial S2  = 4096
#define OFF_SS1  308736            // 16*8*4
#define OFF_SS2  309248            // 16*8*4
#define OFF_SMU  309760            // 16*4
#define OFF_SRS  309824            // 16*4
#define OFF_SCOL 309888            // 16*4
#define OFF_ST   309952            // 16*64*2 f16
#define SMEM_BYTES 312000

__global__ __launch_bounds__(128) void edge_kernel(
        const float* __restrict__ x, const float* __restrict__ hbuf,
        const int* __restrict__ edge_index, const int* __restrict__ edge_attr,
        const float* __restrict__ edgeEmb,
        const float* __restrict__ W1, const float* __restrict__ B1,
        const float* __restrict__ G1, const float* __restrict__ BE1,
        const _Float16* __restrict__ W2h, const float* __restrict__ B2,
        const float* __restrict__ G2, const float* __restrict__ BE2,
        float* __restrict__ sbuf, float* __restrict__ cntbuf) {
    extern __shared__ char smem[];
    float*    sC   = (float*)(smem + OFF_SC);
    float2*   sGB  = (float2*)(smem + OFF_SGB);
    float*    sH   = (float*)(smem + OFF_SH);
    float*    pS1  = (float*)(smem + OFF_PS1);
    float*    pS2  = (float*)(smem + OFF_PS2);
    float*    sS1  = (float*)(smem + OFF_SS1);
    float*    sS2  = (float*)(smem + OFF_SS2);
    float*    sMu  = (float*)(smem + OFF_SMU);
    float*    sRs  = (float*)(smem + OFF_SRS);
    int*      sCol = (int*)  (smem + OFF_SCOL);
    _Float16* sT   = (_Float16*)(smem + OFF_ST);

    const int tid = threadIdx.x;
    const int e   = tid >> 3;          // edge slot 0..15
    const int sub = tid & 7;           // 8 threads per edge
    const int eg  = blockIdx.x * 16 + e;

    // ---- phase 0: stage (g2,be2) transposed, edge gather, t = LN(e@W1+b1) --
    for (int idx = tid; idx < HH; idx += 128) {
        int o = idx & 63, i = idx >> 6;          // idx = i*64 + o
        float2 gb; gb.x = G2[idx]; gb.y = BE2[idx];
        sGB[o * GBPITCH + i] = gb;
    }

    const int r  = edge_index[eg];
    const int c  = edge_index[N_EDGES + eg];
    const int ea = edge_attr[eg];
    float ev[12];
    #pragma unroll
    for (int j = 0; j < 10; ++j) ev[j] = edgeEmb[ea * 10 + j];
    ev[10] = x[r * 3 + 0] - x[c * 3 + 0];
    ev[11] = x[r * 3 + 1] - x[c * 3 + 1];

    float t0[8], ps = 0.f, ps2 = 0.f;
    #pragma unroll
    for (int ii = 0; ii < 8; ++ii) {
        int   i   = sub * 8 + ii;
        float acc = B1[i];
        #pragma unroll
        for (int j = 0; j < 12; ++j) acc += ev[j] * W1[j * H + i];
        t0[ii] = acc;  ps += acc;  ps2 += acc * acc;
    }
    sS1[e * 8 + sub] = ps;  sS2[e * 8 + sub] = ps2;

    #pragma unroll
    for (int ii = 0; ii < 8; ++ii)
        sH[e * H + sub * 8 + ii] = hbuf[r * H + sub * 8 + ii];

    if (sub == 0) { sCol[e] = c; atomicAdd(cntbuf + c, 1.0f); }
    __syncthreads();

    if (tid < 16) {                                   // LN stats over H=64
        float S1 = 0.f, S2 = 0.f;
        #pragma unroll
        for (int j = 0; j < 8; ++j) { S1 += sS1[tid * 8 + j]; S2 += sS2[tid * 8 + j]; }
        float mu  = S1 * (1.f / H);
        float var = S2 * (1.f / H) - mu * mu;
        sMu[tid] = mu;  sRs[tid] = rsqrtf(var + 1e-5f);
    }
    __syncthreads();
    {
        float mu = sMu[e], rs = sRs[e];
        #pragma unroll
        for (int ii = 0; ii < 8; ++ii) {
            int i = sub * 8 + ii;
            sT[e * H + i] = (_Float16)lrelu((t0[ii] - mu) * rs * G1[i] + BE1[i]);
        }
    }
    __syncthreads();

    // ---- phase 1: C(16x4096) = t @ W2 + b2 (WMMA) + in-register LN stats ---
    const int lane = tid & 31;
    const int wv   = __builtin_amdgcn_readfirstlane(tid >> 5);   // scalar wave id
    union Frag { v16h v; uint32_t u[8]; };
    Frag A0, A1;                       // K=0..31 / K=32..63, reused for all tiles
    {
        int M = lane & 15;
        const uint32_t* tp = (const uint32_t*)(sT + M * H);
        #pragma unroll
        for (int v = 0; v < 8; ++v) {
            int Kb = 2 * v + ((lane < 16) ? 0 : 16);
            A0.u[v] = tp[Kb >> 1];
            A1.u[v] = tp[(Kb + 32) >> 1];
        }
    }
    const int rbase = (lane < 16) ? 0 : 8;
    const int ncol  = lane & 15;
    float rS1[8], rS2[8];
    #pragma unroll
    for (int v = 0; v < 8; ++v) { rS1[v] = 0.f; rS2[v] = 0.f; }

    const int ntBase = wv * 64;
    for (int it = 0; it < 32; ++it) {              // 2 tiles / iter, 2 indep chains
        int nt0 = ntBase + it * 2, nt1 = nt0 + 1;
        Frag B00, B01, B10, B11;
        const uint32_t* p00 = (const uint32_t*)W2h + ((size_t)(nt0 * 2 + 0) * 32 + lane) * 8;
        const uint32_t* p01 = (const uint32_t*)W2h + ((size_t)(nt0 * 2 + 1) * 32 + lane) * 8;
        const uint32_t* p10 = (const uint32_t*)W2h + ((size_t)(nt1 * 2 + 0) * 32 + lane) * 8;
        const uint32_t* p11 = (const uint32_t*)W2h + ((size_t)(nt1 * 2 + 1) * 32 + lane) * 8;
        #pragma unroll
        for (int j = 0; j < 8; ++j) {
            B00.u[j] = p00[j]; B01.u[j] = p01[j];
            B10.u[j] = p10[j]; B11.u[j] = p11[j];
        }
        v8f acc0 = {}, acc1 = {};
        acc0 = __builtin_amdgcn_wmma_f32_16x16x32_f16(false, A0.v, false, B00.v,
                                                      (short)0, acc0, false, false);
        acc1 = __builtin_amdgcn_wmma_f32_16x16x32_f16(false, A0.v, false, B10.v,
                                                      (short)0, acc1, false, false);
        acc0 = __builtin_amdgcn_wmma_f32_16x16x32_f16(false, A1.v, false, B01.v,
                                                      (short)0, acc0, false, false);
        acc1 = __builtin_amdgcn_wmma_f32_16x16x32_f16(false, A1.v, false, B11.v,
                                                      (short)0, acc1, false, false);
        float bv0 = B2[nt0 * 16 + ncol];
        float bv1 = B2[nt1 * 16 + ncol];
        int   c0  = nt0 * 16 + ncol;
        int   c1  = nt1 * 16 + ncol;
        #pragma unroll
        for (int v = 0; v < 8; ++v) {
            float v0 = acc0[v] + bv0;
            float v1 = acc1[v] + bv1;
            rS1[v] += v0 + v1;
            rS2[v] += v0 * v0 + v1 * v1;
            sC[(rbase + v) * PITCH + c0] = v0;
            sC[(rbase + v) * PITCH + c1] = v1;
        }
    }
    // dump per-lane row partials, reduce to per-row LN stats
    {
        int pbase = wv * 256 + lane * 8;
        #pragma unroll
        for (int v = 0; v < 8; ++v) { pS1[pbase + v] = rS1[v]; pS2[pbase + v] = rS2[v]; }
    }
    __syncthreads();
    {   // row = tid>>3 : sum 64 partials (4 waves x 16 lanes), 8 per thread
        int row = tid >> 3;
        float S1 = 0.f, S2 = 0.f;
        #pragma unroll
        for (int q = 0; q < 8; ++q) {
            int j   = sub * 8 + q;                 // 0..63
            int wvj = j >> 4;
            int l   = (j & 15) + ((row & 8) ? 16 : 0);
            int idx = wvj * 256 + l * 8 + (row & 7);
            S1 += pS1[idx];  S2 += pS2[idx];
        }
        sS1[row * 8 + sub] = S1;  sS2[row * 8 + sub] = S2;
    }
    __syncthreads();
    if (tid < 16) {
        float S1 = 0.f, S2 = 0.f;
        #pragma unroll
        for (int j = 0; j < 8; ++j) { S1 += sS1[tid * 8 + j]; S2 += sS2[tid * 8 + j]; }
        float mu  = S1 * (1.f / HH);
        float var = S2 * (1.f / HH) - mu * mu;
        sMu[tid] = mu;  sRs[tid] = rsqrtf(var + 1e-5f);
    }
    __syncthreads();

    // ---- phase 3: msg[e,o] = sum_i h[e,i]*lrelu(LN(C)[e,i*64+o]); scatter --
    {
        float mu = sMu[e], rs = sRs[e];
        int   ci = sCol[e];
        const float* rp = sC + e * PITCH;
        float hloc[H];
        #pragma unroll
        for (int i = 0; i < H; ++i) hloc[i] = sH[e * H + i];
        #pragma unroll
        for (int jo = 0; jo < 8; ++jo) {
            int o = sub + jo * 8;
            const float2* gp = sGB + o * GBPITCH;
            float acc = 0.f;
            #pragma unroll 4
            for (int i = 0; i < H; ++i) {
                float2 gb = gp[i];
                float  cv = rp[i * H + o];
                float  w  = lrelu((cv - mu) * rs * gb.x + gb.y);
                acc += hloc[i] * w;
            }
            atomicAdd(sbuf + (size_t)ci * H + o, acc);
        }
    }
}

// ---------------------------------------------------------------------------
// Head: conv = s/cnt + h@root + b ; h += lrelu(conv) ; h += a ; MLP -> scalar
// ---------------------------------------------------------------------------
__global__ void head_kernel(const float* __restrict__ hbuf, const float* __restrict__ abuf,
                            const float* __restrict__ sbuf, const float* __restrict__ cntb,
                            const float* __restrict__ rootW, const float* __restrict__ convB,
                            const float* __restrict__ W0, const float* __restrict__ B0,
                            const float* __restrict__ W1, const float* __restrict__ B1,
                            const float* __restrict__ W2, const float* __restrict__ B2,
                            float* __restrict__ out) {
    int n = blockIdx.x * blockDim.x + threadIdx.x;
    if (n >= P_PRED) return;
    float inv = 1.f / fmaxf(cntb[n], 1.f);
    float hv[H], z[H], z2[H];
    for (int i = 0; i < H; ++i) hv[i] = hbuf[n * H + i];
    for (int i = 0; i < H; ++i) {
        float acc = sbuf[n * H + i] * inv + convB[i];
        for (int k = 0; k < H; ++k) acc += hv[k] * rootW[k * H + i];
        z[i] = hv[i] + lrelu(acc) + abuf[n * H + i];
    }
    for (int i = 0; i < H; ++i) {
        float acc = B0[i];
        for (int k = 0; k < H; ++k) acc += z[k] * W0[k * H + i];
        z2[i] = lrelu(acc);
    }
    for (int i = 0; i < H; ++i) {
        float acc = B1[i];
        for (int k = 0; k < H; ++k) acc += z2[k] * W1[k * H + i];
        z[i] = lrelu(acc);
    }
    float acc = B2[0];
    for (int k = 0; k < H; ++k) acc += z[k] * W2[k];
    out[n] = lrelu(acc);
}

// ---------------------------------------------------------------------------
// workspace layout (bytes):
//   0        W2h f16 packed        524288
//   524288   hbuf  N*64 f32        5120000
//   5644288  abuf  P*64 f32        1280000
//   6924288  sbuf  N*64 f32        5120000
//   12044288 cnt   N    f32        80000     (total ~12.1 MB)
// ---------------------------------------------------------------------------
extern "C" void kernel_launch(void* const* d_in, const int* in_sizes, int n_in,
                              void* d_out, int out_size, void* d_ws, size_t ws_size,
                              hipStream_t stream) {
    (void)in_sizes; (void)n_in; (void)out_size; (void)ws_size;
    const float* x       = (const float*)d_in[0];
    const float* action  = (const float*)d_in[1];
    const float* posW0   = (const float*)d_in[2];
    const float* posB0   = (const float*)d_in[3];
    const float* posW1   = (const float*)d_in[4];
    const float* posB1   = (const float*)d_in[5];
    const float* entEmb  = (const float*)d_in[6];
    const float* edgeEmb = (const float*)d_in[7];
    const float* actW    = (const float*)d_in[8];
    const float* actB    = (const float*)d_in[9];
    const float* nn1W    = (const float*)d_in[10];
    const float* nn1B    = (const float*)d_in[11];
    const float* ln1G    = (const float*)d_in[12];
    const float* ln1B    = (const float*)d_in[13];
    const float* nn2W    = (const float*)d_in[14];
    const float* nn2B    = (const float*)d_in[15];
    const float* ln2G    = (const float*)d_in[16];
    const float* ln2B    = (const float*)d_in[17];
    const float* rootW   = (const float*)d_in[18];
    const float* convB   = (const float*)d_in[19];
    const float* netW0   = (const float*)d_in[20];
    const float* netB0   = (const float*)d_in[21];
    const float* netW1   = (const float*)d_in[22];
    const float* netB1   = (const float*)d_in[23];
    const float* netW2   = (const float*)d_in[24];
    const float* netB2   = (const float*)d_in[25];
    const int* edgeIndex = (const int*)d_in[26];
    const int* edgeAttr  = (const int*)d_in[27];
    const int* mask      = (const int*)d_in[28];
    float* out = (float*)d_out;

    char* ws = (char*)d_ws;
    _Float16* W2h  = (_Float16*)(ws + 0);
    float*    hbuf = (float*)(ws + 524288);
    float*    abuf = (float*)(ws + 5644288);
    float*    sbuf = (float*)(ws + 6924288);
    float*    cntb = (float*)(ws + 12044288);

    hipMemsetAsync(sbuf, 0, (size_t)N_NODES * H * sizeof(float), stream);
    hipMemsetAsync(cntb, 0, (size_t)N_NODES * sizeof(float), stream);

    pack_w2_kernel<<<1024, 256, 0, stream>>>(nn2W, W2h);
    node_kernel<<<(N_NODES + 255) / 256, 256, 0, stream>>>(
        x, action, mask, posW0, posB0, posW1, posB1, entEmb, actW, actB, hbuf, abuf);

    hipFuncSetAttribute((const void*)edge_kernel,
                        hipFuncAttributeMaxDynamicSharedMemorySize, SMEM_BYTES);
    edge_kernel<<<N_EDGES / 16, 128, SMEM_BYTES, stream>>>(
        x, hbuf, edgeIndex, edgeAttr, edgeEmb, nn1W, nn1B, ln1G, ln1B,
        W2h, nn2B, ln2G, ln2B, sbuf, cntb);

    head_kernel<<<(P_PRED + 127) / 128, 128, 0, stream>>>(
        hbuf, abuf, sbuf, cntb, rootW, convB,
        netW0, netB0, netW1, netB1, netW2, netB2, out);
}
